// SegmentDeactivation_55155970015966
// MI455X (gfx1250) — matrix operand order repaired and checked by reading, not compiled, not run on hardware
//
#include <hip/hip_runtime.h>

// ---------------------------------------------------------------------------
// KAN layer on MI455X (gfx1250): one fused bf16 WMMA GEMM (split-K).
//   out = [silu(x) | x | dense_bases] @ [base_w | mask*a | (1-mask)*w']^T + c
// K_TOT = 512 + 512 + 512*36 = 19456
// ---------------------------------------------------------------------------

#define BATCH 1024
#define IN_F  512
#define OUT_F 512
#define NCOEF 35          // G + k
#define CPAD  36          // padded coef group (keeps K % 32 == 0)
#define K_TOT (2*IN_F + IN_F*CPAD)   // 19456 = 608 * 32
#define SPLITK 4
#define KSPLIT (K_TOT / SPLITK)      // 4864 = 152 * 32

typedef __bf16 bf16;
typedef __attribute__((ext_vector_type(16))) __bf16 v16bf;
typedef __attribute__((ext_vector_type(8)))  float  v8f;
typedef __attribute__((ext_vector_type(4)))  unsigned int v4u;

union FragU { v4u u4[2]; v16bf v; };

// workspace layout (bytes)
#define R_OFF    0                                        // 512*19456*2  = 19,922,944
#define L_OFF    ((size_t)OUT_F * K_TOT * 2)              // + 1024*19456*2
#define BIAS_OFF (L_OFF + (size_t)BATCH * K_TOT * 2)      // + 512*4
#define PART_OFF (BIAS_OFF + 2048)                        // + 4*1024*512*4 = 8 MB

// ---------------------------------------------------------------------------
// Build right matrix R[o, K_TOT] (bf16):
//   cols [0,512)        : base_weight[o,i]
//   cols [512,1024)     : mask ? a_lin[o,i] : 0
//   cols [1024 + 36*i+c): mask ? 0 : spline_weight[o,i,c]*scaler[o,i]  (c<35), 0 pad
// a_lin from closed-form spline endpoints on the uniform grid:
//   y(-1) = (w0+4w1+w2)/6 , y(+1) = (w32+4w33+w34)/6 , a = (y(+1)-y(-1))/2
// ---------------------------------------------------------------------------
__global__ __launch_bounds__(256) void kan_prep_weights(
    const float* __restrict__ base_w, const float* __restrict__ spline_w,
    const float* __restrict__ scaler, const unsigned char* __restrict__ mask,
    bf16* __restrict__ R)
{
    int t = blockIdx.x * blockDim.x + threadIdx.x;
    if (t >= OUT_F * IN_F) return;
    int o = t >> 9;
    int i = t & 511;

    float s  = scaler[t];
    bool  dm = mask[t] != 0;
    const float* w = spline_w + (size_t)t * NCOEF;

    bf16* row = R + (size_t)o * K_TOT;
    row[i] = (bf16)base_w[t];

    float wp[CPAD];
    #pragma unroll
    for (int c = 0; c < NCOEF; ++c) wp[c] = w[c] * s;
    wp[NCOEF] = 0.f;

    float y_start = (wp[0]  + 4.f*wp[1]  + wp[2])  * (1.f/6.f);
    float y_end   = (wp[32] + 4.f*wp[33] + wp[34]) * (1.f/6.f);
    float a_lin   = (y_end - y_start) * 0.5f;          // dx = 2
    row[IN_F + i] = (bf16)(dm ? a_lin : 0.f);

    union { bf16 h[CPAD]; unsigned long long q[9]; } sp;
    #pragma unroll
    for (int c = 0; c < CPAD; ++c) sp.h[c] = (bf16)(dm ? 0.f : wp[c]);
    unsigned long long* dst =
        (unsigned long long*)(row + 2*IN_F + (size_t)i*CPAD);   // 8B aligned (72*i)
    #pragma unroll
    for (int q = 0; q < 9; ++q) dst[q] = sp.q[q];
}

// ---------------------------------------------------------------------------
// bias[o] = sum_i mask[o,i] * b_lin[o,i],  b_lin = y_start + a_lin
// ---------------------------------------------------------------------------
__global__ __launch_bounds__(256) void kan_bias(
    const float* __restrict__ spline_w, const float* __restrict__ scaler,
    const unsigned char* __restrict__ mask, float* __restrict__ bias)
{
    __shared__ float red[256];
    int o = blockIdx.x;
    float acc = 0.f;
    for (int i = threadIdx.x; i < IN_F; i += 256) {
        int t = o * IN_F + i;
        if (mask[t]) {
            float s = scaler[t];
            const float* w = spline_w + (size_t)t * NCOEF;
            float y_start = (w[0]  + 4.f*w[1]  + w[2])  * s * (1.f/6.f);
            float y_end   = (w[32] + 4.f*w[33] + w[34]) * s * (1.f/6.f);
            float a_lin   = (y_end - y_start) * 0.5f;
            acc += y_start + a_lin;                    // b = y_start - a*(-1)
        }
    }
    red[threadIdx.x] = acc;
    __syncthreads();
    for (int s = 128; s > 0; s >>= 1) {
        if (threadIdx.x < s) red[threadIdx.x] += red[threadIdx.x + s];
        __syncthreads();
    }
    if (threadIdx.x == 0) bias[o] = red[0];
}

// ---------------------------------------------------------------------------
// Build left matrix L[b, K_TOT] (bf16): silu(x), x, and closed-form cubic
// B-spline bases on the uniform extended grid (knot0 = -1.1875, h = 1/16).
// Only bases m-3..m are nonzero for x in cell m; outside the knots -> all 0.
// ---------------------------------------------------------------------------
__global__ __launch_bounds__(256) void kan_build_L(
    const float* __restrict__ X, bf16* __restrict__ L)
{
    int t = blockIdx.x * blockDim.x + threadIdx.x;
    if (t >= BATCH * IN_F) return;
    int b = t >> 9;
    int i = t & 511;

    float x = X[t];
    bf16* row = L + (size_t)b * K_TOT;

    float sig = 1.f / (1.f + __expf(-x));
    row[i]        = (bf16)(x * sig);   // silu for base path
    row[IN_F + i] = (bf16)x;           // raw x for masked linear path

    union { bf16 h[CPAD]; unsigned long long q[9]; } sp;
    #pragma unroll
    for (int q = 0; q < 9; ++q) sp.q[q] = 0ull;

    float tp = (x + 1.1875f) * 16.f;   // position in knot units
    int   m  = (int)floorf(tp);
    if (m >= 0 && m < 38) {            // inside the extended grid
        float u  = tp - (float)m;
        float om = 1.f - u;
        float u2 = u*u, u3 = u2*u;
        float bb0 = om*om*om * (1.f/6.f);                       // basis m-3
        float bb1 = (3.f*u3 - 6.f*u2 + 4.f) * (1.f/6.f);        // basis m-2
        float bb2 = (-3.f*u3 + 3.f*u2 + 3.f*u + 1.f)*(1.f/6.f); // basis m-1
        float bb3 = u3 * (1.f/6.f);                             // basis m
        float bb[4] = { bb0, bb1, bb2, bb3 };
        #pragma unroll
        for (int j = 0; j < 4; ++j) {
            int idx = m - 3 + j;
            if (idx >= 0 && idx < NCOEF) sp.h[idx] = (bf16)bb[j];
        }
    }
    unsigned long long* dst =
        (unsigned long long*)(row + 2*IN_F + (size_t)i*CPAD);
    #pragma unroll
    for (int q = 0; q < 9; ++q) dst[q] = sp.q[q];
}

// ---------------------------------------------------------------------------
// Split-K WMMA GEMM: part[s] += L[1024,Ks] x R[512,Ks]^T
// Block = 128 threads (4 waves), tile 128(M) x 64(N); each wave owns a
// 32x64 sub-tile = 2x4 v_wmma_f32_16x16x32_bf16 per K-step
// (8 WMMAs per 12 b128 loads; B fragments shared by all 4 waves via WGP$).
// Fragment loads use contiguous K order, the same for A and B:
//   A (16x32): lane%16 = row; lanes 0-15 hold K 0-15, lanes 16-31 K 16-31
//   B (32x16): lane = column; lanes 0-15 hold K 0-15, lanes 16-31 K 16-31
// ---------------------------------------------------------------------------
__global__ __launch_bounds__(128) void kan_wmma_gemm(
    const bf16* __restrict__ L, const bf16* __restrict__ R,
    float* __restrict__ part)
{
    const int lane   = threadIdx.x & 31;
    const int wave   = threadIdx.x >> 5;             // 0..3 -> M sub-tile
    const int m0     = blockIdx.x * 128 + wave * 32;
    const int n0     = blockIdx.y * 64;
    const int kBeg   = blockIdx.z * KSPLIT;
    const int kEnd   = kBeg + KSPLIT;

    const int half   = lane >> 4;       // 0: lanes 0-15, 1: lanes 16-31
    const int lane15 = lane & 15;
    const int aK0 = half * 16;          // A per-lane k start
    const int bK0 = half * 16;          // B per-lane k start

    const bf16* aRow0 = L + (size_t)(m0 + lane15) * K_TOT;
    const bf16* aRow1 = aRow0 + (size_t)16 * K_TOT;
    const bf16* bRow0 = R + (size_t)(n0 + lane15) * K_TOT;
    const bf16* bRow1 = bRow0 + (size_t)16 * K_TOT;
    const bf16* bRow2 = bRow0 + (size_t)32 * K_TOT;
    const bf16* bRow3 = bRow0 + (size_t)48 * K_TOT;

    v8f acc00 = {}, acc01 = {}, acc02 = {}, acc03 = {};
    v8f acc10 = {}, acc11 = {}, acc12 = {}, acc13 = {};

    for (int kk = kBeg; kk < kEnd; kk += 32) {
        // prefetch ~16 iterations ahead (speculative: OOB silently dropped)
        __builtin_prefetch(aRow0 + kk + 512, 0, 1);
        __builtin_prefetch(aRow1 + kk + 512, 0, 1);
        __builtin_prefetch(bRow0 + kk + 512, 0, 1);
        __builtin_prefetch(bRow2 + kk + 512, 0, 1);

        FragU a0, a1, b0, b1, b2, b3;
        const v4u* pa0 = (const v4u*)(aRow0 + kk + aK0);
        const v4u* pa1 = (const v4u*)(aRow1 + kk + aK0);
        const v4u* pb0 = (const v4u*)(bRow0 + kk + bK0);
        const v4u* pb1 = (const v4u*)(bRow1 + kk + bK0);
        const v4u* pb2 = (const v4u*)(bRow2 + kk + bK0);
        const v4u* pb3 = (const v4u*)(bRow3 + kk + bK0);
        a0.u4[0] = pa0[0]; a0.u4[1] = pa0[1];   // 16 contiguous K
        a1.u4[0] = pa1[0]; a1.u4[1] = pa1[1];
        b0.u4[0] = pb0[0]; b0.u4[1] = pb0[1];   // 16 contiguous K
        b1.u4[0] = pb1[0]; b1.u4[1] = pb1[1];
        b2.u4[0] = pb2[0]; b2.u4[1] = pb2[1];
        b3.u4[0] = pb3[0]; b3.u4[1] = pb3[1];

        acc00 = __builtin_amdgcn_wmma_f32_16x16x32_bf16(false, a0.v, false, b0.v,
                                                        (short)0, acc00, false, false);
        acc01 = __builtin_amdgcn_wmma_f32_16x16x32_bf16(false, a0.v, false, b1.v,
                                                        (short)0, acc01, false, false);
        acc02 = __builtin_amdgcn_wmma_f32_16x16x32_bf16(false, a0.v, false, b2.v,
                                                        (short)0, acc02, false, false);
        acc03 = __builtin_amdgcn_wmma_f32_16x16x32_bf16(false, a0.v, false, b3.v,
                                                        (short)0, acc03, false, false);
        acc10 = __builtin_amdgcn_wmma_f32_16x16x32_bf16(false, a1.v, false, b0.v,
                                                        (short)0, acc10, false, false);
        acc11 = __builtin_amdgcn_wmma_f32_16x16x32_bf16(false, a1.v, false, b1.v,
                                                        (short)0, acc11, false, false);
        acc12 = __builtin_amdgcn_wmma_f32_16x16x32_bf16(false, a1.v, false, b2.v,
                                                        (short)0, acc12, false, false);
        acc13 = __builtin_amdgcn_wmma_f32_16x16x32_bf16(false, a1.v, false, b3.v,
                                                        (short)0, acc13, false, false);
    }

    // C/D layout: lane15 = N column; VGPR r -> M row = r + 8*half
    float* pOut = part + (size_t)blockIdx.z * BATCH * OUT_F;
    const int rowOff = half * 8;
    v8f accs[2][4] = { { acc00, acc01, acc02, acc03 },
                       { acc10, acc11, acc12, acc13 } };
    #pragma unroll
    for (int mt = 0; mt < 2; ++mt) {
        #pragma unroll
        for (int nt = 0; nt < 4; ++nt) {
            int col = n0 + nt*16 + lane15;
            int rowBase = m0 + mt*16 + rowOff;
            #pragma unroll
            for (int r = 0; r < 8; ++r)
                pOut[(size_t)(rowBase + r) * OUT_F + col] = accs[mt][nt][r];
        }
    }
}

// ---------------------------------------------------------------------------
// out[b,o] = sum_s part[s][b,o] + bias[o]
// ---------------------------------------------------------------------------
__global__ __launch_bounds__(256) void kan_reduce(
    const float* __restrict__ part, const float* __restrict__ bias,
    float* __restrict__ out)
{
    int t = blockIdx.x * blockDim.x + threadIdx.x;
    if (t >= BATCH * OUT_F) return;
    float s = bias[t & (OUT_F - 1)];
    #pragma unroll
    for (int k = 0; k < SPLITK; ++k)
        s += part[(size_t)k * BATCH * OUT_F + t];
    out[t] = s;
}

// ---------------------------------------------------------------------------
extern "C" void kernel_launch(void* const* d_in, const int* in_sizes, int n_in,
                              void* d_out, int out_size, void* d_ws, size_t ws_size,
                              hipStream_t stream)
{
    const float* x        = (const float*)d_in[0];
    const float* base_w   = (const float*)d_in[1];
    const float* spline_w = (const float*)d_in[2];
    const float* scaler   = (const float*)d_in[3];
    // d_in[4] = grid: uniform extended knots, closed form used instead
    const unsigned char* mask = (const unsigned char*)d_in[5];
    float* out = (float*)d_out;

    char* ws = (char*)d_ws;
    bf16*  R    = (bf16*)(ws + R_OFF);
    bf16*  Lm   = (bf16*)(ws + L_OFF);
    float* bias = (float*)(ws + BIAS_OFF);
    float* part = (float*)(ws + PART_OFF);

    kan_prep_weights<<<(OUT_F*IN_F + 255)/256, 256, 0, stream>>>(
        base_w, spline_w, scaler, mask, R);
    kan_bias<<<OUT_F, 256, 0, stream>>>(spline_w, scaler, mask, bias);
    kan_build_L<<<(BATCH*IN_F + 255)/256, 256, 0, stream>>>(x, Lm);

    dim3 grid(BATCH/128, OUT_F/64, SPLITK);   // 8 x 8 x 4 = 256 workgroups
    kan_wmma_gemm<<<grid, 128, 0, stream>>>(Lm, R, part);

    kan_reduce<<<(BATCH*OUT_F + 255)/256, 256, 0, stream>>>(part, bias, out);
}
